// CriticNetwork_22711787061383
// MI455X (gfx1250) — matrix operand order repaired and checked by reading, not compiled, not run on hardware
//
#include <hip/hip_runtime.h>
#include <hip/hip_bf16.h>

#define NN 50000
#define NE 800000
#define NG 64

typedef __attribute__((ext_vector_type(16))) _Float16 v16h;
typedef __attribute__((ext_vector_type(8)))  float    v8f;

__device__ __forceinline__ float lrelu(float x){ return x > 0.f ? x : 0.2f*x; }
__device__ __forceinline__ float eluf (float x){ return x > 0.f ? x : (__expf(x)-1.f); }

// order-preserving float<->uint encoding for atomic max
__device__ __forceinline__ unsigned int fenc(float f){
  unsigned int b = __float_as_uint(f);
  return (b & 0x80000000u) ? ~b : (b | 0x80000000u);
}
__device__ __forceinline__ float fdec(unsigned int e){
  return __uint_as_float((e & 0x80000000u) ? (e & 0x7FFFFFFFu) : ~e);
}

// ---------------------------------------------------------------------------
// Prep: contract weights with attention vectors into tiny coefficient tables,
// build Qsrc/Qdst = Ws2 . a_{src,dst}2 (256x4), and Ws2^T in f16 for WMMA.
// coef layout: [0:4) c_src1, [4:8) c_dst1, [8:16) d1[j*4+h], [16:24) e2[j*4+h],
//              [24:27) wd_src, [27:30) wd_dst
// ---------------------------------------------------------------------------
__global__ void k_prep(const float* __restrict__ Ws1,const float* __restrict__ a_src1,
                       const float* __restrict__ a_dst1,const float* __restrict__ We1,
                       const float* __restrict__ ae1,const float* __restrict__ Ws2,
                       const float* __restrict__ a_src2,const float* __restrict__ a_dst2,
                       const float* __restrict__ We2,const float* __restrict__ ae2,
                       const float* __restrict__ Wd,const float* __restrict__ ad_src,
                       const float* __restrict__ ad_dst,
                       float* __restrict__ coef,float* __restrict__ Qsrc,
                       float* __restrict__ Qdst,_Float16* __restrict__ Bt){
  int t = threadIdx.x;                 // 256 threads
  if (t < 4){ float s=0.f; for(int c=0;c<64;++c) s += Ws1[t*64+c]*a_src1[t*64+c]; coef[t]=s; }
  else if (t < 8){ int h=t-4; float s=0.f; for(int c=0;c<64;++c) s += Ws1[h*64+c]*a_dst1[h*64+c]; coef[t]=s; }
  else if (t < 16){ int i=t-8,j=i>>2,h=i&3; float s=0.f;
    for(int c=0;c<64;++c) s += We1[j*256+h*64+c]*ae1[h*64+c]; coef[t]=s; }
  else if (t < 24){ int i=t-16,j=i>>2,h=i&3; float s=0.f;
    for(int c=0;c<64;++c) s += We2[j*256+h*64+c]*ae2[h*64+c]; coef[t]=s; }
  else if (t < 27){ int k=t-24; float s=0.f; for(int c=0;c<64;++c) s += Wd[k*64+c]*ad_src[c]; coef[t]=s; }
  else if (t < 30){ int k=t-27; float s=0.f; for(int c=0;c<64;++c) s += Wd[k*64+c]*ad_dst[c]; coef[t]=s; }
  // Qsrc/Qdst: one k per thread
  {
    int k = t;
    for(int h=0;h<4;++h){
      float ss=0.f, sd=0.f;
      for(int c=0;c<64;++c){
        float w = Ws2[k*256 + h*64 + c];
        ss += w*a_src2[h*64+c]; sd += w*a_dst2[h*64+c];
      }
      Qsrc[k*4+h]=ss; Qdst[k*4+h]=sd;
    }
  }
  // Bt[n][k] = Ws2[k][n] in f16
  for (int i = t; i < 256*256; i += 256){
    int ncol = i >> 8, k = i & 255;
    Bt[i] = (_Float16)Ws2[k*256 + ncol];
  }
}

// ---------------------------------------------------------------------------
// Layer-1 (4 heads) + dynamic (1 head) edge logits — recomputed each pass.
// ---------------------------------------------------------------------------
__device__ __forceinline__ void edge1_logits(const float* __restrict__ x,
    const float* __restrict__ ea,const float* __restrict__ coef,
    int s,int d,int e,float al[5]){
  float xs0=x[s*4+0], xd0=x[d*4+0];
  float e0=ea[e*2+0], e1=ea[e*2+1];
  #pragma unroll
  for(int h=0;h<4;++h)
    al[h]=lrelu(xs0*coef[h] + xd0*coef[4+h] + e0*coef[8+h] + e1*coef[12+h]);
  float ss = x[s*4+1]*coef[24] + x[s*4+2]*coef[25] + x[s*4+3]*coef[26];
  float sd = x[d*4+1]*coef[27] + x[d*4+2]*coef[28] + x[d*4+3]*coef[29];
  al[4]=lrelu(ss+sd);
}

__global__ void k_e1_max(const int* __restrict__ src,const int* __restrict__ dst,
                         const float* __restrict__ x,const float* __restrict__ ea,
                         const float* __restrict__ coef,unsigned int* __restrict__ m1){
  int e = blockIdx.x*blockDim.x + threadIdx.x; if (e>=NE) return;
  int s=src[e], d=dst[e]; float al[5]; edge1_logits(x,ea,coef,s,d,e,al);
  #pragma unroll
  for(int h=0;h<5;++h) atomicMax(&m1[d*5+h], fenc(al[h]));
}

__global__ void k_e1_sum(const int* __restrict__ src,const int* __restrict__ dst,
                         const float* __restrict__ x,const float* __restrict__ ea,
                         const float* __restrict__ coef,const unsigned int* __restrict__ m1,
                         float* __restrict__ den1){
  int e = blockIdx.x*blockDim.x + threadIdx.x; if (e>=NE) return;
  int s=src[e], d=dst[e]; float al[5]; edge1_logits(x,ea,coef,s,d,e,al);
  #pragma unroll
  for(int h=0;h<5;++h) atomicAdd(&den1[d*5+h], __expf(al[h]-fdec(m1[d*5+h])));
}

__global__ void k_e1_acc(const int* __restrict__ src,const int* __restrict__ dst,
                         const float* __restrict__ x,const float* __restrict__ ea,
                         const float* __restrict__ coef,const unsigned int* __restrict__ m1,
                         const float* __restrict__ den1,
                         float* __restrict__ w1sum,float* __restrict__ xagg){
  int e = blockIdx.x*blockDim.x + threadIdx.x; if (e>=NE) return;
  int s=src[e], d=dst[e]; float al[5]; edge1_logits(x,ea,coef,s,d,e,al);
  float a[5];
  #pragma unroll
  for(int h=0;h<5;++h) a[h]=__expf(al[h]-fdec(m1[d*5+h]))/(den1[d*5+h]+1e-16f);
  float xs0=x[s*4+0];
  #pragma unroll
  for(int h=0;h<4;++h) atomicAdd(&w1sum[d*4+h], a[h]*xs0);
  #pragma unroll
  for(int k=0;k<3;++k) atomicAdd(&xagg[d*3+k], a[4]*x[s*4+1+k]);
}

// ---------------------------------------------------------------------------
// Per-node: h1 = ELU(w1sum⊗Ws1 + bs1) -> f16; layer-2 logits s2 = h1.Q;
// h_final base = xagg@Wd + bd + bs2.    one block (64 threads) per node
// ---------------------------------------------------------------------------
__global__ __launch_bounds__(64) void k_node1(
    const float* __restrict__ w1sum,const float* __restrict__ xagg,
    const float* __restrict__ Ws1,const float* __restrict__ bs1,
    const float* __restrict__ Qsrc,const float* __restrict__ Qdst,
    const float* __restrict__ Wd,const float* __restrict__ bd,
    const float* __restrict__ bs2,
    _Float16* __restrict__ h1,float* __restrict__ s2src,float* __restrict__ s2dst,
    float* __restrict__ hfin){
  int n = blockIdx.x, c = threadIdx.x;
  __shared__ float red[8*64];
  float ps[4]={0,0,0,0}, pd[4]={0,0,0,0};
  #pragma unroll
  for(int h=0;h<4;++h){
    int k = h*64+c;
    float v = eluf(w1sum[n*4+h]*Ws1[k] + bs1[k]);
    h1[(size_t)n*256+k] = (_Float16)v;
    #pragma unroll
    for(int q=0;q<4;++q){ ps[q]+=v*Qsrc[k*4+q]; pd[q]+=v*Qdst[k*4+q]; }
  }
  #pragma unroll
  for(int q=0;q<4;++q){ red[q*64+c]=ps[q]; red[(4+q)*64+c]=pd[q]; }
  __syncthreads();
  for(int sft=32; sft>0; sft>>=1){
    if(c<sft){
      #pragma unroll
      for(int r=0;r<8;++r) red[r*64+c]+=red[r*64+c+sft];
    }
    __syncthreads();
  }
  if(c<4) s2src[n*4+c]=red[c*64];
  else if(c<8) s2dst[n*4+(c-4)]=red[(c)*64];
  float hv = xagg[n*3+0]*Wd[c] + xagg[n*3+1]*Wd[64+c] + xagg[n*3+2]*Wd[128+c]
           + bd[c] + bs2[c];
  hfin[(size_t)n*64+c] = hv;
}

// ---------------------------------------------------------------------------
// WMMA GEMM: h2(50000x256) = h1(50000x256,f16) @ Ws2(256x256) using Bt (f16, N-major).
// Block = 256 thr (8 waves). Each block: one 16-row strip; wave w does cols [32w,32w+32).
// ---------------------------------------------------------------------------
__global__ __launch_bounds__(256) void k_gemm(const _Float16* __restrict__ A,
                                              const _Float16* __restrict__ Bt,
                                              float* __restrict__ C,int M){
  __shared__ _Float16 As[16*256];
  int m0 = blockIdx.x*16;
  { // cooperative A-strip load (16 halfs per thread, 128-bit vectors)
    int r = threadIdx.x >> 4, seg = (threadIdx.x & 15)*16, row = m0 + r;
    if (row < M){
      const uint4* s = (const uint4*)(A + (size_t)row*256 + seg);
      uint4* t = (uint4*)&As[r*256+seg];
      t[0]=s[0]; t[1]=s[1];
    } else {
      for(int i=0;i<16;++i) As[r*256+seg+i]=(_Float16)0.f;
    }
  }
  __syncthreads();
  int wave = threadIdx.x>>5, lane = threadIdx.x&31;
  int r  = lane & 15;            // A row within tile / C column index
  int kbA = (lane>>4)*16;        // A: lanes 16-31 carry K+16 group
  int kbB = (lane>>4)*16;        // B: lanes 16-31 carry K+16 group
  int n0 = wave*32;
  v8f acc0 = {}; v8f acc1 = {};
  #pragma unroll
  for(int ks=0; ks<8; ++ks){
    int ko = ks*32;
    v16h a, b0, b1;
    const _Float16* ap  = &As[r*256 + ko + kbA];
    const _Float16* bp0 = Bt + (size_t)(n0      + r)*256 + ko + kbB;
    const _Float16* bp1 = Bt + (size_t)(n0 + 16 + r)*256 + ko + kbB;
    #pragma unroll
    for(int i=0;i<8;++i){ a[i]=ap[i]; a[8+i]=ap[8+i]; }
    #pragma unroll
    for(int i=0;i<16;++i){ b0[i]=bp0[i]; b1[i]=bp1[i]; }
    acc0 = __builtin_amdgcn_wmma_f32_16x16x32_f16(false,a,false,b0,(short)0,acc0,false,false);
    acc1 = __builtin_amdgcn_wmma_f32_16x16x32_f16(false,a,false,b1,(short)0,acc1,false,false);
  }
  #pragma unroll
  for(int i=0;i<8;++i){
    int row = m0 + (lane>>4)*8 + i;
    if (row < M){
      C[(size_t)row*256 + n0 + r]      = acc0[i];
      C[(size_t)row*256 + n0 + 16 + r] = acc1[i];
    }
  }
}

// ---------------------------------------------------------------------------
// Layer-2 edge softmax (4 heads) + message scatter of h2 (mean over heads)
// ---------------------------------------------------------------------------
__device__ __forceinline__ void edge2_logits(const float* __restrict__ ss,
    const float* __restrict__ sd,const float* __restrict__ ea,
    const float* __restrict__ coef,int s,int d,int e,float al[4]){
  float e0=ea[e*2+0], e1=ea[e*2+1];
  #pragma unroll
  for(int h=0;h<4;++h)
    al[h]=lrelu(ss[s*4+h] + sd[d*4+h] + e0*coef[16+h] + e1*coef[20+h]);
}

__global__ void k_e2_max(const int* __restrict__ src,const int* __restrict__ dst,
                         const float* __restrict__ ss,const float* __restrict__ sd,
                         const float* __restrict__ ea,const float* __restrict__ coef,
                         unsigned int* __restrict__ m2){
  int e = blockIdx.x*blockDim.x + threadIdx.x; if(e>=NE) return;
  int s=src[e], d=dst[e]; float al[4]; edge2_logits(ss,sd,ea,coef,s,d,e,al);
  #pragma unroll
  for(int h=0;h<4;++h) atomicMax(&m2[d*4+h], fenc(al[h]));
}

__global__ void k_e2_sum(const int* __restrict__ src,const int* __restrict__ dst,
                         const float* __restrict__ ss,const float* __restrict__ sd,
                         const float* __restrict__ ea,const float* __restrict__ coef,
                         const unsigned int* __restrict__ m2,float* __restrict__ den2){
  int e = blockIdx.x*blockDim.x + threadIdx.x; if(e>=NE) return;
  int s=src[e], d=dst[e]; float al[4]; edge2_logits(ss,sd,ea,coef,s,d,e,al);
  #pragma unroll
  for(int h=0;h<4;++h) atomicAdd(&den2[d*4+h], __expf(al[h]-fdec(m2[d*4+h])));
}

__global__ __launch_bounds__(64) void k_e2_acc(
    const int* __restrict__ src,const int* __restrict__ dst,
    const float* __restrict__ ss,const float* __restrict__ sd,
    const float* __restrict__ ea,const float* __restrict__ coef,
    const unsigned int* __restrict__ m2,const float* __restrict__ den2,
    const float* __restrict__ h2,float* __restrict__ hfin){
  int e = blockIdx.x, c = threadIdx.x;
  int s=src[e], d=dst[e]; float al[4]; edge2_logits(ss,sd,ea,coef,s,d,e,al);
  float acc=0.f;
  #pragma unroll
  for(int h=0;h<4;++h){
    float a = __expf(al[h]-fdec(m2[d*4+h]))/(den2[d*4+h]+1e-16f);
    acc += a * h2[(size_t)s*256 + h*64 + c];
  }
  atomicAdd(&hfin[(size_t)d*64+c], 0.25f*acc);
}

// ---------------------------------------------------------------------------
// Pooling (batch is sorted): per-block local run-accumulate, flush on boundary
// ---------------------------------------------------------------------------
__global__ __launch_bounds__(64) void k_pool(const float* __restrict__ hfin,
    const int* __restrict__ batch,float* __restrict__ g,float* __restrict__ counts){
  int c = threadIdx.x;
  int start = blockIdx.x*256, end = min(start+256, NN);
  if (start>=end) return;
  int cur = batch[start];
  float acc=0.f, cnt=0.f;
  for(int i=start;i<end;++i){
    int b = batch[i];
    if (b != cur){
      atomicAdd(&g[cur*64+c], acc);
      if(c==0) atomicAdd(&counts[cur], cnt);
      acc=0.f; cnt=0.f; cur=b;
    }
    acc += hfin[(size_t)i*64+c];
    cnt += 1.f;
  }
  atomicAdd(&g[cur*64+c], acc);
  if(c==0) atomicAdd(&counts[cur], cnt);
}

__global__ __launch_bounds__(64) void k_mlp(const float* __restrict__ g,
    const float* __restrict__ counts,const float* __restrict__ Wv1,
    const float* __restrict__ bv1,const float* __restrict__ Wv2,
    const float* __restrict__ bv2,float* __restrict__ out){
  __shared__ float gs[NG*64];
  int t = threadIdx.x;                      // graph id
  float cnt = counts[t]; cnt = cnt>0.f?cnt:1.f;
  for(int k=0;k<64;++k) gs[t*64+k] = g[t*64+k]/cnt;
  __syncthreads();
  float acc=0.f;
  for(int j=0;j<64;++j){
    float s = bv1[j];
    for(int k=0;k<64;++k) s += gs[t*64+k]*Wv1[k*64+j];
    s = s>0.f?s:0.f;
    acc += s*Wv2[j];
  }
  out[t] = acc + bv2[0];
}

// ---------------------------------------------------------------------------
extern "C" void kernel_launch(void* const* d_in, const int* in_sizes, int n_in,
                              void* d_out, int out_size, void* d_ws, size_t ws_size,
                              hipStream_t stream){
  const float* x       = (const float*)d_in[0];
  const float* eattr   = (const float*)d_in[1];
  const float* Ws1     = (const float*)d_in[2];
  const float* as_src1 = (const float*)d_in[3];
  const float* as_dst1 = (const float*)d_in[4];
  const float* We1     = (const float*)d_in[5];
  const float* ae1     = (const float*)d_in[6];
  const float* bs1     = (const float*)d_in[7];
  const float* Ws2     = (const float*)d_in[8];
  const float* as_src2 = (const float*)d_in[9];
  const float* as_dst2 = (const float*)d_in[10];
  const float* We2     = (const float*)d_in[11];
  const float* ae2     = (const float*)d_in[12];
  const float* bs2     = (const float*)d_in[13];
  const float* Wd      = (const float*)d_in[14];
  const float* ad_src  = (const float*)d_in[15];
  const float* ad_dst  = (const float*)d_in[16];
  const float* bd      = (const float*)d_in[17];
  const float* Wv1     = (const float*)d_in[18];
  const float* bv1     = (const float*)d_in[19];
  const float* Wv2     = (const float*)d_in[20];
  const float* bv2     = (const float*)d_in[21];
  const int*   ei      = (const int*)d_in[22];
  const int*   batch   = (const int*)d_in[23];
  const int* src = ei, *dst = ei + NE;
  float* out = (float*)d_out;

  // workspace carve-up (256B aligned)
  char* base = (char*)d_ws; size_t o = 0;
  auto carve = [&](size_t bytes)->char*{ char* p = base + o; o = (o + bytes + 255) & ~(size_t)255; return p; };
  float*        coef  = (float*)       carve(32*4);
  float*        Qsrc  = (float*)       carve(256*4*4);
  float*        Qdst  = (float*)       carve(256*4*4);
  _Float16*     Bt    = (_Float16*)    carve(256*256*2);
  _Float16*     h1    = (_Float16*)    carve((size_t)NN*256*2);
  float*        h2    = (float*)       carve((size_t)NN*256*4);
  float*        hfin  = (float*)       carve((size_t)NN*64*4);
  char*         z0    =                carve(0);          // start of zero-init block
  unsigned int* m1    = (unsigned int*)carve((size_t)NN*5*4);
  float*        den1  = (float*)       carve((size_t)NN*5*4);
  float*        w1sum = (float*)       carve((size_t)NN*4*4);
  float*        xagg  = (float*)       carve((size_t)NN*3*4);
  float*        s2src = (float*)       carve((size_t)NN*4*4);
  float*        s2dst = (float*)       carve((size_t)NN*4*4);
  unsigned int* m2    = (unsigned int*)carve((size_t)NN*4*4);
  float*        den2  = (float*)       carve((size_t)NN*4*4);
  float*        g     = (float*)       carve((size_t)NG*64*4);
  float*        cnts  = (float*)       carve((size_t)NG*4);
  size_t zbytes = (size_t)((base + o) - z0);

  (void)hipMemsetAsync(z0, 0, zbytes, stream);  // zero: maxima(enc 0 = -max), dens, accums, pool

  k_prep<<<1,256,0,stream>>>(Ws1,as_src1,as_dst1,We1,ae1,Ws2,as_src2,as_dst2,
                             We2,ae2,Wd,ad_src,ad_dst,coef,Qsrc,Qdst,Bt);

  int egrid = (NE + 255)/256;
  k_e1_max<<<egrid,256,0,stream>>>(src,dst,x,eattr,coef,m1);
  k_e1_sum<<<egrid,256,0,stream>>>(src,dst,x,eattr,coef,m1,den1);
  k_e1_acc<<<egrid,256,0,stream>>>(src,dst,x,eattr,coef,m1,den1,w1sum,xagg);

  k_node1<<<NN,64,0,stream>>>(w1sum,xagg,Ws1,bs1,Qsrc,Qdst,Wd,bd,bs2,
                              h1,s2src,s2dst,hfin);

  k_gemm<<<(NN+15)/16,256,0,stream>>>(h1,Bt,h2,NN);

  k_e2_max<<<egrid,256,0,stream>>>(src,dst,s2src,s2dst,eattr,coef,m2);
  k_e2_sum<<<egrid,256,0,stream>>>(src,dst,s2src,s2dst,eattr,coef,m2,den2);
  k_e2_acc<<<NE,64,0,stream>>>(src,dst,s2src,s2dst,eattr,coef,m2,den2,h2,hfin);

  k_pool<<<(NN+255)/256,64,0,stream>>>(hfin,batch,g,cnts);
  k_mlp<<<1,64,0,stream>>>(g,cnts,Wv1,bv1,Wv2,bv2,out);
}
